// MultiHeadAttention_80865644249683
// MI455X (gfx1250) — matrix operand order repaired and checked by reading, not compiled, not run on hardware
//
#include <hip/hip_runtime.h>
#include <hip/hip_bf16.h>
#include <math.h>

typedef _Float16 f16;
typedef __attribute__((ext_vector_type(16))) _Float16 v16h;
typedef __attribute__((ext_vector_type(8)))  _Float16 v8h;
typedef __attribute__((ext_vector_type(8)))  float    v8f;
typedef __attribute__((ext_vector_type(4)))  unsigned int v4u;
typedef __attribute__((ext_vector_type(8)))  int v8i;
typedef __attribute__((ext_vector_type(4)))  int v4i;

#define B_   2
#define S_   2048
#define E_   1024
#define H_   16
#define D_   64
#define N3_  3072   // 3*E
#define BS_  4096   // B*S

#if __has_include(<hip/amd_detail/amd_gfx1250_TDM.h>)
#define TDM_6ARG 1
#else
#define TDM_6ARG 0
#endif

static __device__ __forceinline__ v16h combine8(v8h lo, v8h hi) {
  return __builtin_shufflevector(lo, hi, 0,1,2,3,4,5,6,7,8,9,10,11,12,13,14,15);
}

// A-fragment for V_WMMA_F32_16X16X32_F16 (16x32, K window starting at a_row_k0):
// lane L: row m = L&15; elems 0..7  = A[m][(L>>4)*16 + 0..7]
//                       elems 8..15 = A[m][(L>>4)*16 + 8 + 0..7]
static __device__ __forceinline__ v16h load_afrag(const f16* a_row_k0, int half) {
  v8h lo = *(const v8h*)(a_row_k0 + half * 16);
  v8h hi = *(const v8h*)(a_row_k0 + half * 16 + 8);
  return combine8(lo, hi);
}

// B-fragment (32x16): lane L: col n = L&15; elem i = B[(L>>4)*16 + i][n].
// With B^T stored row-major (N x K) this is 16 contiguous f16 (32B).
static __device__ __forceinline__ v16h load_bfrag(const f16* bt_row_k0, int half) {
  return *(const v16h*)(bt_row_k0 + half * 16);
}

static __device__ __forceinline__ v8f wmma16(v16h a, v16h b, v8f c) {
  return __builtin_amdgcn_wmma_f32_16x16x32_f16(false, a, false, b,
                                                (short)0, c, false, false);
}

// CDNA5 async copy global -> LDS (16B per lane), tracked by ASYNCcnt.
static __device__ __forceinline__ void async_b128_to_lds(unsigned lds_off,
                                                         const void* gaddr) {
  asm volatile("global_load_async_to_lds_b128 %0, %1, off"
               :: "v"(lds_off), "v"((unsigned long long)(size_t)gaddr)
               : "memory");
}
static __device__ __forceinline__ void wait_asynccnt0() {
  asm volatile("s_wait_asynccnt 0x0" ::: "memory");
}

// CDNA5 Tensor Data Mover: one descriptor moves a 2-D strided tile to LDS.
// D# per ISA 8.3/8.4: group0 = {count|flags, lds_addr, global_addr[31:0],
// global_addr[56:32]|type=2}; group1 packs data_size/dims/strides.
static __device__ __forceinline__ void tdm_load_2d(unsigned lds_bytes,
                                                   const void* gaddr,
                                                   int tile_d0, int tile_d1,
                                                   int d0_stride,
                                                   int tensor_d0, int tensor_d1) {
  const unsigned long long ga = (unsigned long long)(size_t)gaddr;
  v4u g0;
  g0.x = 1u;                                                // count=1 (valid D#)
  g0.y = lds_bytes;                                         // lds_addr
  g0.z = (unsigned)(ga & 0xffffffffu);                      // global_addr[31:0]
  g0.w = (unsigned)((ga >> 32) & 0x01ffffffu) | (2u << 30); // [56:32] | type=2
  v8i g1;
  g1[0] = 0x00010000;                                       // data_size=1 -> 2B
  g1[1] = (tensor_d0 & 0xffff) << 16;                       // tensor_dim0[15:0]
  g1[2] = ((tensor_d0 >> 16) & 0xffff) | ((tensor_d1 & 0xffff) << 16);
  g1[3] = ((tensor_d1 >> 16) & 0xffff) | ((tile_d0 & 0xffff) << 16);
  g1[4] = (tile_d1 & 0xffff);                               // tile_dim1, tile_dim2=0
  g1[5] = d0_stride;                                        // tensor_dim0_stride
  g1[6] = 0;
  g1[7] = 0;
  const v4i gz = {0, 0, 0, 0};
#if TDM_6ARG
  const v8i gz8 = {0, 0, 0, 0, 0, 0, 0, 0};
  __builtin_amdgcn_tensor_load_to_lds(g0, g1, gz, gz, gz8, 0);
#else
  __builtin_amdgcn_tensor_load_to_lds(g0, g1, gz, gz, 0);
#endif
}

// ---------------- precision-convert / transpose kernels ----------------

__global__ void k_cast_f16(const float* __restrict__ src, f16* __restrict__ dst, int n) {
  int i = blockIdx.x * blockDim.x + threadIdx.x;
  if (i < n) dst[i] = (f16)src[i];
}

// src: K x N (row-major f32) -> dst: N x K (row-major f16)
__global__ void k_transpose_f16(const float* __restrict__ src, f16* __restrict__ dst,
                                int K, int N) {
  int i = blockIdx.x * blockDim.x + threadIdx.x;
  if (i >= K * N) return;
  int n = i / K, k = i - n * K;
  dst[i] = (f16)src[(size_t)k * N + n];
}

// ---------------- fused QKV projection GEMM (TDM-staged) ----------------
// Block = 4 waves = 64x64 output tile. Wave w owns rows [w*16, w*16+16) x 64 cols
// (4 wmma accumulators, A-frag reused 4x). B^T panel (64 x 32k = 4KB) is double-
// buffered in LDS via the Tensor Data Mover (TENSORcnt): wave 0 issues one
// tensor_load_to_lds per K-chunk, waits s_wait_tensorcnt 0, then the workgroup
// barrier publishes the panel to all 4 waves.
__global__ void __launch_bounds__(128)
k_gemm_qkv(const f16* __restrict__ xh,      // BS x E
           const f16* __restrict__ wT,      // N3 x E  (Wqkv^T)
           const float* __restrict__ bias,  // N3
           f16* __restrict__ qb, f16* __restrict__ kb, f16* __restrict__ vtb) {
  __shared__ f16 Bt[2][64 * 32];            // 2 x 4KB double buffer
  const int tid = threadIdx.x;
  const int lane = tid & 31, wave = tid >> 5;
  const int i = blockIdx.x & 63;            // M block (64 rows)
  const int j = blockIdx.x >> 6;            // N block (64 cols), 0..47
  const int half = lane >> 4, nn = lane & 15;

  const f16* arow = xh + (size_t)(i * 64 + wave * 16 + nn) * E_;

  auto issue = [&](int buf, int k0) {
    if (wave == 0)
      tdm_load_2d((unsigned)(size_t)&Bt[buf][0],
                  wT + (size_t)(j * 64) * E_ + k0,
                  /*tile_d0=*/32, /*tile_d1=*/64,
                  /*d0_stride=*/E_, /*tensor_d0=*/E_, /*tensor_d1=*/N3_);
  };

  v8f acc[4] = {};
  issue(0, 0);
  int cur = 0;
  for (int k0 = 0; k0 < E_; k0 += 32) {
    if (wave == 0) __builtin_amdgcn_s_wait_tensorcnt(0);
    __syncthreads();                        // panel ready; prev compute done
    if (k0 + 32 < E_) issue(cur ^ 1, k0 + 32);
    __builtin_prefetch(arow + k0 + 128, 0, 1);   // global_prefetch_b8 (A stream)
    const v16h a = load_afrag(arow + k0, half);
    const f16* bb = &Bt[cur][0];
#pragma unroll
    for (int t = 0; t < 4; ++t)
      acc[t] = wmma16(a, *(const v16h*)(bb + (t * 16 + nn) * 32 + half * 16), acc[t]);
    cur ^= 1;
  }

#pragma unroll
  for (int t = 0; t < 4; ++t) {
    const int col = j * 64 + t * 16 + nn;
    const int h = col / (3 * D_);
    const int rem = col - h * 3 * D_;
    const int part = rem / D_;
    const int d = rem - part * D_;
    const float bv = bias[col];
#pragma unroll
    for (int r = 0; r < 8; ++r) {
      const int row = i * 64 + wave * 16 + r + 8 * half;  // global token index
      const int bidx = row >> 11;                         // / S_
      const int s = row & (S_ - 1);
      const int bh = bidx * H_ + h;
      const f16 v = (f16)(acc[t][r] + bv);
      if (part == 0)      qb[((size_t)bh * S_ + s) * D_ + d] = v;
      else if (part == 1) kb[((size_t)bh * S_ + s) * D_ + d] = v;
      else                vtb[((size_t)bh * D_ + d) * S_ + s] = v;
    }
  }
}

// ---------------- causal flash attention ----------------
// one wave per 16-row Q block; streams 32-key blocks with online softmax.
// Waves have different trip counts -> per-wave LDS slice + s_wait_dscnt (no barrier).
__global__ void __launch_bounds__(128)
k_flash(const f16* __restrict__ qb, const f16* __restrict__ kbuf,
        const f16* __restrict__ vtb, f16* __restrict__ attn) {
  __shared__ f16 ldsP[4][16 * 32];                // per-wave P-tile bounce
  const int lane = threadIdx.x & 31;
  const int wave = threadIdx.x >> 5;
  const int task = blockIdx.x * 4 + wave;         // qblk*32 + bh
  const int bh = task & 31;
  const int qi = (task >> 5) * 16;
  const int half = lane >> 4, nn = lane & 15;

  const f16* Qb = qb + (size_t)bh * S_ * D_;
  const f16* Kb = kbuf + (size_t)bh * S_ * D_;
  const f16* Vt = vtb + (size_t)bh * D_ * S_;
  f16* myP = ldsP[wave];

  const f16* qrow = Qb + (size_t)(qi + nn) * D_;
  const v16h qa0 = load_afrag(qrow, half);        // K = 0..31 of HD
  const v16h qa1 = load_afrag(qrow + 32, half);   // K = 32..63

  v8f o0 = {}, o1 = {}, o2 = {}, o3 = {};
  float mrow[8], lrow[8];
#pragma unroll
  for (int r = 0; r < 8; ++r) { mrow[r] = -INFINITY; lrow[r] = 0.f; }

  const float sc = 0.125f;                        // 1/sqrt(64)

  for (int kb0 = 0; kb0 <= qi + 15; kb0 += 32) {
    // S(16x32) = Q(16x64) @ K^T(64x32): two 16x16 N-tiles, K=64 -> 2 wmma each
    const f16* k0row = Kb + (size_t)(kb0 + nn) * D_;
    const f16* k1row = Kb + (size_t)(kb0 + 16 + nn) * D_;
    v8f s0 = {}, s1 = {};
    s0 = wmma16(qa0, load_bfrag(k0row, half), s0);
    s0 = wmma16(qa1, load_bfrag(k0row + 32, half), s0);
    s1 = wmma16(qa0, load_bfrag(k1row, half), s1);
    s1 = wmma16(qa1, load_bfrag(k1row + 32, half), s1);

    const bool pm = (kb0 + 31 > qi);              // partial (diagonal) block?
    float p0[8], p1[8];
#pragma unroll
    for (int r = 0; r < 8; ++r) {
      const int qg = qi + r + 8 * half;           // D-layout row
      float a = s0[r] * sc;
      float b = s1[r] * sc;
      if (pm) {
        if (kb0 + nn > qg)      a += -1e9f;
        if (kb0 + 16 + nn > qg) b += -1e9f;
      }
      float mx = fmaxf(a, b);
#pragma unroll
      for (int off = 1; off < 16; off <<= 1)
        mx = fmaxf(mx, __shfl_xor(mx, off, 32));  // row max over 16 lanes of half
      const float mnew = fmaxf(mrow[r], mx);
      const float alpha = __expf(mrow[r] - mnew);
      mrow[r] = mnew;
      const float pa = __expf(a - mnew);
      const float pb = __expf(b - mnew);
      float rs = pa + pb;
#pragma unroll
      for (int off = 1; off < 16; off <<= 1)
        rs += __shfl_xor(rs, off, 32);
      lrow[r] = lrow[r] * alpha + rs;
      o0[r] *= alpha; o1[r] *= alpha; o2[r] *= alpha; o3[r] *= alpha;
      p0[r] = pa; p1[r] = pb;
    }
    // D-layout -> LDS (row-major 16x32) -> reload as A-frag
#pragma unroll
    for (int r = 0; r < 8; ++r) {
      const int m = r + 8 * half;
      myP[m * 32 + nn]      = (f16)p0[r];
      myP[m * 32 + 16 + nn] = (f16)p1[r];
    }
    __asm__ volatile("s_wait_dscnt 0" ::: "memory");
    const v16h pfrag = load_afrag(myP + nn * 32, half);
    // O(16x64) += P(16x32) @ V(32x64): V^T rows are contiguous in keys
    o0 = wmma16(pfrag, *(const v16h*)(Vt + ((size_t)(nn))      * S_ + kb0 + half * 16), o0);
    o1 = wmma16(pfrag, *(const v16h*)(Vt + ((size_t)(16 + nn)) * S_ + kb0 + half * 16), o1);
    o2 = wmma16(pfrag, *(const v16h*)(Vt + ((size_t)(32 + nn)) * S_ + kb0 + half * 16), o2);
    o3 = wmma16(pfrag, *(const v16h*)(Vt + ((size_t)(48 + nn)) * S_ + kb0 + half * 16), o3);
  }

#pragma unroll
  for (int r = 0; r < 8; ++r) {
    const int srow = qi + r + 8 * half;
    const float inv = 1.0f / lrow[r];
    const size_t base = ((size_t)(bh >> 4) * S_ + srow) * E_ + (size_t)(bh & 15) * D_;
    attn[base + nn]      = (f16)(o0[r] * inv);
    attn[base + 16 + nn] = (f16)(o1[r] * inv);
    attn[base + 32 + nn] = (f16)(o2[r] * inv);
    attn[base + 48 + nn] = (f16)(o3[r] * inv);
  }
}

// ---------------- output projection GEMM (f32 out, async-LDS staged) --------
__global__ void __launch_bounds__(128)
k_gemm_out(const f16* __restrict__ ah,   // BS x E
           const f16* __restrict__ woT,  // E x E (Wo^T)
           const float* __restrict__ bo, // E
           float* __restrict__ out) {    // BS x E
  __shared__ f16 Bt[2][64 * 32];
  const int tid = threadIdx.x;
  const int lane = tid & 31, wave = tid >> 5;
  const int i = blockIdx.x & 63;          // M block
  const int j = blockIdx.x >> 6;          // N block, 0..15
  const int half = lane >> 4, nn = lane & 15;

  const f16* arow = ah + (size_t)(i * 64 + wave * 16 + nn) * E_;

  auto issue = [&](int buf, int k0) {
#pragma unroll
    for (int it = 0; it < 2; ++it) {
      const int q = it * 128 + tid;       // 256 x 16B transfers = 4KB
      const int row = q >> 2, seg = q & 3;
      const f16* g = woT + (size_t)(j * 64 + row) * E_ + k0 + seg * 8;
      const unsigned l = (unsigned)(size_t)&Bt[buf][row * 32 + seg * 8];
      async_b128_to_lds(l, g);
    }
  };

  v8f acc[4] = {};
  issue(0, 0);
  int cur = 0;
  for (int k0 = 0; k0 < E_; k0 += 32) {
    wait_asynccnt0();
    __syncthreads();
    if (k0 + 32 < E_) issue(cur ^ 1, k0 + 32);
    __builtin_prefetch(arow + k0 + 128, 0, 1);
    const v16h a = load_afrag(arow + k0, half);
    const f16* bb = &Bt[cur][0];
#pragma unroll
    for (int t = 0; t < 4; ++t)
      acc[t] = wmma16(a, *(const v16h*)(bb + (t * 16 + nn) * 32 + half * 16), acc[t]);
    cur ^= 1;
  }

#pragma unroll
  for (int t = 0; t < 4; ++t) {
    const int col = j * 64 + t * 16 + nn;
    const float bv = bo[col];
#pragma unroll
    for (int r = 0; r < 8; ++r) {
      const int row = i * 64 + wave * 16 + r + 8 * half;
      out[(size_t)row * E_ + col] = acc[t][r] + bv;
    }
  }
}

// ---------------- host launch ----------------
extern "C" void kernel_launch(void* const* d_in, const int* in_sizes, int n_in,
                              void* d_out, int out_size, void* d_ws, size_t ws_size,
                              hipStream_t stream) {
  const float* x    = (const float*)d_in[0];
  const float* Wqkv = (const float*)d_in[1];
  const float* bqkv = (const float*)d_in[2];
  const float* Wo   = (const float*)d_in[3];
  const float* bo   = (const float*)d_in[4];
  // d_in[5] is the additive causal mask: synthesized analytically, never read.

  char* p = (char*)d_ws;
  f16* xh    = (f16*)p; p += (size_t)BS_ * E_ * sizeof(f16);   // 8 MB
  f16* wqkvT = (f16*)p; p += (size_t)N3_ * E_ * sizeof(f16);   // 6 MB
  f16* woT   = (f16*)p; p += (size_t)E_ * E_ * sizeof(f16);    // 2 MB
  f16* qb    = (f16*)p; p += (size_t)B_ * H_ * S_ * D_ * sizeof(f16); // 8 MB
  f16* kb    = (f16*)p; p += (size_t)B_ * H_ * S_ * D_ * sizeof(f16); // 8 MB
  f16* vtb   = (f16*)p; p += (size_t)B_ * H_ * S_ * D_ * sizeof(f16); // 8 MB
  f16* attnh = (f16*)p; p += (size_t)BS_ * E_ * sizeof(f16);   // 8 MB

  k_cast_f16<<<(BS_ * E_ + 255) / 256, 256, 0, stream>>>(x, xh, BS_ * E_);
  k_transpose_f16<<<(E_ * N3_ + 255) / 256, 256, 0, stream>>>(Wqkv, wqkvT, E_, N3_);
  k_transpose_f16<<<(E_ * E_ + 255) / 256, 256, 0, stream>>>(Wo, woT, E_, E_);

  // 64 x 48 blocks of 64x64 output tiles, 4 waves (16x64 each) per block
  k_gemm_qkv<<<64 * 48, 128, 0, stream>>>(xh, wqkvT, bqkv, qb, kb, vtb);
  // 2*16*128 q-block tasks, 4 waves/block
  k_flash<<<(B_ * H_ * (S_ / 16)) / 4, 128, 0, stream>>>(qb, kb, vtb, attnh);
  // 64 x 16 blocks
  k_gemm_out<<<64 * 16, 128, 0, stream>>>(attnh, woT, bo, (float*)d_out);
}
